// KANLayer_13228499272146
// MI455X (gfx1250) — compile-verified, matrix operand order checked by reading, not run on hardware
//
#include <hip/hip_runtime.h>
#include <hip/hip_bf16.h>
#include <stddef.h>

// ---------------------------------------------------------------------------
// KAN layer: out[b,o] = sum_d c0*xn^3 + c1*xn^2 + c2*xn  (+ bias[o])
// xn = per-column min-max normalized x.   B=4096, D=1024, O=1024.
// Folded into a K=3072 bf16 GEMM using v_wmma_f32_16x16x32_bf16 (f32 acc),
// with A/B pre-swizzled into the WMMA 16-bit per-lane layout so the GEMM
// inner loop is pure b128 vector loads + WMMA.
// ---------------------------------------------------------------------------

typedef __bf16  v16bf __attribute__((ext_vector_type(16)));
typedef float   v8f   __attribute__((ext_vector_type(8)));

#define BATCH 4096
#define DIM   1024
#define ODIM  1024

// ---- workspace layout (bytes) ----
#define WS_PMIN   0                      // 16*1024 f32
#define WS_PMAX   65536                  // 16*1024 f32
#define WS_XMIN   131072                 // 1024 f32
#define WS_XMAX   135168                 // 1024 f32
#define WS_BIAS   139264                 // 1024 f32
#define WS_APACK  262144                 // 3*256*32 tiles * 1KB = 25165824 B
#define WS_BPACK  (262144 + 25165824)    // 3*64*32 tiles * 1KB  =  6291456 B

__device__ __forceinline__ unsigned short f32_to_bf16(float f) {
    union { float f; unsigned int u; } v; v.f = f;
    unsigned int u = v.u;
    u += 0x7FFFu + ((u >> 16) & 1u);     // round-to-nearest-even
    return (unsigned short)(u >> 16);
}

// ISA 7.12.2, 16-bit operand layout (wave32):
// lanes 0-15: VGPR v<4 holds K=2v+h ; v>=4 holds K=16+2(v-4)+h
// lanes 16-31: same +8.
__device__ __forceinline__ int wmma16_k(int v, int h, int lane) {
    int base = (lane >> 4) * 8;
    return (v < 4 ? 2 * v + h : 16 + 2 * (v - 4) + h) + base;
}

// ---------------- pass 1: column min/max, deterministic 2-pass -------------
__global__ void __launch_bounds__(256) kan_minmax_part(const float* __restrict__ x,
                                                       float* __restrict__ pmin,
                                                       float* __restrict__ pmax) {
    int d  = blockIdx.x * 256 + threadIdx.x;   // 0..1023
    int rb = blockIdx.y;                       // 0..15 (256 rows each)
    const float* p = x + (size_t)rb * 256 * DIM + d;
    float mn = p[0], mx = p[0];
    #pragma unroll 8
    for (int i = 1; i < 256; ++i) {
        float v = p[(size_t)i * DIM];
        mn = fminf(mn, v);
        mx = fmaxf(mx, v);
    }
    pmin[rb * DIM + d] = mn;
    pmax[rb * DIM + d] = mx;
}

__global__ void __launch_bounds__(256) kan_minmax_final(const float* __restrict__ pmin,
                                                        const float* __restrict__ pmax,
                                                        float* __restrict__ xmin,
                                                        float* __restrict__ xmax) {
    int d = blockIdx.x * 256 + threadIdx.x;
    float mn = pmin[d], mx = pmax[d];
    #pragma unroll
    for (int i = 1; i < 16; ++i) {
        mn = fminf(mn, pmin[i * DIM + d]);
        mx = fmaxf(mx, pmax[i * DIM + d]);
    }
    xmin[d] = mn;
    xmax[d] = mx;
}

// ---------------- B-prep: sum spline points, bf16, WMMA swizzle ------------
// Tile = (p, oTile 16 cols, dTile 32 rows) = 32 lanes * 16 bf16 = 1 KB.
// Tile index: (p*64 + ot)*32 + dt.  8 tiles per 256-thread block.
__global__ void __launch_bounds__(256) kan_bprep(const float* __restrict__ coeffs,
                                                 unsigned short* __restrict__ Bpk) {
    int t    = blockIdx.x * 256 + threadIdx.x;
    int lane = t & 31;
    int tile = t >> 5;            // 0..6143
    int dt   = tile & 31;
    int ot   = (tile >> 5) & 63;
    int p    = tile >> 11;        // 0..2  -> coeff index k = p
    int o    = ot * 16 + (lane & 15);

    unsigned short outv[16] __attribute__((aligned(16)));
    #pragma unroll
    for (int v = 0; v < 8; ++v) {
        #pragma unroll
        for (int h = 0; h < 2; ++h) {
            int d = dt * 32 + wmma16_k(v, h, lane);
            const float* cp = coeffs + ((size_t)o * DIM + d) * 16 + p;
            float s = cp[0] + cp[4] + cp[8] + cp[12];    // sum over spline points
            outv[2 * v + h] = f32_to_bf16(s);
        }
    }
    unsigned short* dst = Bpk + (size_t)tile * 512 + lane * 16;
    ((uint4*)dst)[0] = ((const uint4*)outv)[0];
    ((uint4*)dst)[1] = ((const uint4*)outv)[1];
}

// ---------------- bias: sum_{d,s} coeffs[o,d,s,3] --------------------------
// One block per output column; deterministic LDS tree reduction.
__global__ void __launch_bounds__(256) kan_bias(const float* __restrict__ coeffs,
                                                float* __restrict__ bias) {
    __shared__ float red[256];
    const int o = blockIdx.x;                  // 0..1023
    const float* cp = coeffs + (size_t)o * (DIM * 16) + 3;
    float s = 0.f;
    #pragma unroll 4
    for (int i = threadIdx.x; i < DIM * 4; i += 256) s += cp[(size_t)i * 4];
    red[threadIdx.x] = s;
    __syncthreads();
    #pragma unroll
    for (int w = 128; w > 0; w >>= 1) {
        if ((int)threadIdx.x < w) red[threadIdx.x] += red[threadIdx.x + w];
        __syncthreads();
    }
    if (threadIdx.x == 0) bias[o] = red[0];
}

// ---------------- A-prep: normalize, powers, bf16, WMMA swizzle ------------
// Tile = (p, rowTile 16, dTile 32).  Tile index: (p*256 + rt)*32 + dt.
__global__ void __launch_bounds__(256) kan_aprep(const float* __restrict__ x,
                                                 const float* __restrict__ xmin,
                                                 const float* __restrict__ xmax,
                                                 unsigned short* __restrict__ Apk) {
    int t    = blockIdx.x * 256 + threadIdx.x;
    int lane = t & 31;
    int tile = t >> 5;            // 0..8191
    int dt   = tile & 31;
    int rt   = tile >> 5;         // 0..255
    int row  = rt * 16 + (lane & 15);

    unsigned short outv[3][16] __attribute__((aligned(16)));
    #pragma unroll
    for (int v = 0; v < 8; ++v) {
        #pragma unroll
        for (int h = 0; h < 2; ++h) {
            int d  = dt * 32 + wmma16_k(v, h, lane);
            float mn = xmin[d], mx = xmax[d];
            float xn = (x[(size_t)row * DIM + d] - mn) / (mx - mn);
            float x2 = xn * xn;
            float x3 = x2 * xn;
            outv[0][2 * v + h] = f32_to_bf16(x3);   // pairs with c0
            outv[1][2 * v + h] = f32_to_bf16(x2);   // pairs with c1
            outv[2][2 * v + h] = f32_to_bf16(xn);   // pairs with c2
        }
    }
    #pragma unroll
    for (int p = 0; p < 3; ++p) {
        unsigned short* dst =
            Apk + (size_t)((p * 256 + rt) * 32 + dt) * 512 + lane * 16;
        ((uint4*)dst)[0] = ((const uint4*)outv[p])[0];
        ((uint4*)dst)[1] = ((const uint4*)outv[p])[1];
    }
}

// ---------------- GEMM: 8 waves/block, wave -> 32 rows x 64 cols -----------
// Block covers 256 rows x 64 cols.  Per K-step: 2 A loads + 4 B loads feed
// 8 WMMAs (each B operand reused by two row tiles back-to-back).
__global__ void __launch_bounds__(256) kan_gemm(const unsigned short* __restrict__ Apk,
                                                const unsigned short* __restrict__ Bpk,
                                                const float* __restrict__ bias,
                                                float* __restrict__ out) {
    const int lane = threadIdx.x & 31;
    const int wave = threadIdx.x >> 5;
    const int rt0    = blockIdx.x * 16 + wave * 2;  // first of 2 row tiles
    const int otBase = blockIdx.y * 4;              // first of 4 o-tiles

    const v16bf* Av = (const v16bf*)Apk;            // 32 v16bf per 1KB tile
    const v16bf* Bv = (const v16bf*)Bpk;

    v8f acc[2][4];
    #pragma unroll
    for (int i = 0; i < 2; ++i)
        #pragma unroll
        for (int j = 0; j < 4; ++j) acc[i][j] = (v8f)0.f;

    for (int p = 0; p < 3; ++p) {
        const size_t abase0 = (size_t)((p * 256 + rt0) * 32) * 32 + lane;
        const size_t abase1 = abase0 + 1024;        // next row tile (+32 tiles)
        const size_t bbase  = (size_t)((p * 64 + otBase) * 32) * 32 + lane;
        #pragma unroll 4
        for (int dt = 0; dt < 32; ++dt) {
            v16bf a0 = Av[abase0 + (size_t)dt * 32];
            v16bf a1 = Av[abase1 + (size_t)dt * 32];
            #pragma unroll
            for (int j = 0; j < 4; ++j) {
                v16bf b = Bv[bbase + (size_t)j * 1024 + (size_t)dt * 32];
                acc[0][j] = __builtin_amdgcn_wmma_f32_16x16x32_bf16(
                    false, a0, false, b, (short)0, acc[0][j], false, false);
                acc[1][j] = __builtin_amdgcn_wmma_f32_16x16x32_bf16(
                    false, a1, false, b, (short)0, acc[1][j], false, false);
            }
        }
    }

    // C/D layout: lanes 0-15 -> N=lane, M=vgpr ; lanes 16-31 -> N=lane-16, M=8+vgpr
    const int n     = lane & 15;
    const int mOffs = (lane >> 4) * 8;
    #pragma unroll
    for (int i = 0; i < 2; ++i) {
        const int rowBase = (rt0 + i) * 16 + mOffs;
        #pragma unroll
        for (int j = 0; j < 4; ++j) {
            const int col = blockIdx.y * 64 + j * 16 + n;
            const float bv = bias[col];
            #pragma unroll
            for (int r = 0; r < 8; ++r) {
                out[(size_t)(rowBase + r) * ODIM + col] = acc[i][j][r] + bv;
            }
        }
    }
}

// ---------------------------------------------------------------------------
extern "C" void kernel_launch(void* const* d_in, const int* in_sizes, int n_in,
                              void* d_out, int out_size, void* d_ws, size_t ws_size,
                              hipStream_t stream) {
    (void)in_sizes; (void)n_in; (void)out_size; (void)ws_size;
    const float* x      = (const float*)d_in[0];   // [4096,1024]
    const float* coeffs = (const float*)d_in[1];   // [1024,1024,4,4]
    float* out = (float*)d_out;

    char* ws = (char*)d_ws;
    float* pmin = (float*)(ws + WS_PMIN);
    float* pmax = (float*)(ws + WS_PMAX);
    float* xmin = (float*)(ws + WS_XMIN);
    float* xmax = (float*)(ws + WS_XMAX);
    float* bias = (float*)(ws + WS_BIAS);
    unsigned short* Apk = (unsigned short*)(ws + WS_APACK);
    unsigned short* Bpk = (unsigned short*)(ws + WS_BPACK);

    kan_minmax_part <<<dim3(4, 16), 256, 0, stream>>>(x, pmin, pmax);
    kan_minmax_final<<<4,          256, 0, stream>>>(pmin, pmax, xmin, xmax);
    kan_bprep       <<<768,        256, 0, stream>>>(coeffs, Bpk);   // 6144 tiles
    kan_bias        <<<1024,       256, 0, stream>>>(coeffs, bias);
    kan_aprep       <<<1024,       256, 0, stream>>>(x, xmin, xmax, Apk); // 8192 tiles
    kan_gemm        <<<dim3(16, 16), 256, 0, stream>>>(Apk, Bpk, bias, out);
}
